// MultiHeadAttention_64321430225040
// MI455X (gfx1250) — compile-verified, matrix operand order checked by reading, not run on hardware
//
#include <hip/hip_runtime.h>
#include <hip/hip_bf16.h>

// ---------------------------------------------------------------------------
// MHA forward for MI455X (gfx1250), all matmuls via v_wmma_f32_16x16x32_bf16.
//   B=2, S=2048, D=1024, H=16, dk=64.  ~52 GFLOP executed (~69 dense), ~50MB unique traffic.
// GEMMs are register-blocked: one wave owns a 32x64 output block
// (2x4 WMMA tiles), so each K-step does 8 WMMAs on 6 fragment loads.
// Workspace layout (48 MB total):
//   xb[4096x1024]bf16, wq/wk/wv/wo[1024x1024]bf16,
//   Qh,Kh [B,H,S,dk]bf16 (RoPE applied), Vt [B,H,dk,S]bf16, Ao [4096x1024]bf16
// ---------------------------------------------------------------------------

#define BATCH  2
#define SEQL   2048
#define DMODEL 1024
#define HEADS  16
#define DK     64

typedef __attribute__((ext_vector_type(16))) __bf16 v16bf;
typedef __attribute__((ext_vector_type(8)))  __bf16 v8bf;
typedef __attribute__((ext_vector_type(8)))  float  v8f;

// Load one 16-element half-pair of a 16-bit WMMA A/B fragment:
// elements 0..7  = p[0..7]       (K = kbase..kbase+7)
// elements 8..15 = p[16..23]     (K = 16+kbase..16+kbase+7)
__device__ __forceinline__ v16bf load_frag(const __bf16* p) {
    union { v16bf v; v8bf h[2]; } u;
    u.h[0] = *(const v8bf*)(p);
    u.h[1] = *(const v8bf*)(p + 16);
    return u.v;
}

__device__ __forceinline__ v8f wmma_bf16(v16bf a, v16bf b, v8f c) {
    return __builtin_amdgcn_wmma_f32_16x16x32_bf16(
        /*neg_a=*/false, a, /*neg_b=*/false, b,
        /*c_mod=*/(short)0, c, /*reuse_a=*/false, /*reuse_b=*/false);
}

// ---------------------------------------------------------------------------
__global__ __launch_bounds__(256) void f32_to_bf16_kernel(
    const float* __restrict__ in, __bf16* __restrict__ out, int n) {
    int i = blockIdx.x * blockDim.x + threadIdx.x;
    if (i < n) out[i] = (__bf16)in[i];
}

// ---------------------------------------------------------------------------
// Register-blocked GEMM: C[m][n] = sum_k A[m][k] * W[n][k]   (i.e. x @ W^T)
// One wave owns a 32x64 block = 2 (M) x 4 (N) WMMA tiles; K-loop in steps
// of 32 does 8 WMMAs per 6 fragment loads (A reused 4x, B reused 2x).
// MODE 0: RoPE epilogue -> bf16 store to [B,H,S,dk]      (Q and K)
// MODE 1: transpose     -> bf16 store to [B,H,dk,S]      (V)
// MODE 2: plain f32     -> [M,N]                         (output proj)
template<int MODE>
__global__ __launch_bounds__(256) void gemm_blocked_kernel(
    const __bf16* __restrict__ A, const __bf16* __restrict__ W,
    void* __restrict__ dst, const int* __restrict__ pos,
    int M, int N, int K)
{
    const int lane  = threadIdx.x & 31;
    const int wave  = threadIdx.x >> 5;
    const int stN   = N >> 6;                       // supertiles along N (64 wide)
    const int st    = blockIdx.x * 8 + wave;
    if (st >= (M >> 5) * stN) return;               // wave-uniform

    const int m0 = (st / stN) << 5;                 // 32-row block
    const int n0 = (st % stN) << 6;                 // 64-col block
    const int kb = (lane & 16) ? 8 : 0;
    const int ln = lane & 15;
    const int hi = lane >> 4;

    const __bf16* arow[2];
    const __bf16* wrow[4];
    #pragma unroll
    for (int mi = 0; mi < 2; ++mi) arow[mi] = A + (size_t)(m0 + mi * 16 + ln) * K;
    #pragma unroll
    for (int ni = 0; ni < 4; ++ni) wrow[ni] = W + (size_t)(n0 + ni * 16 + ln) * K;

    v8f acc[2][4] = {};
    for (int k0 = 0; k0 < K; k0 += 32) {
        v16bf af[2], bf[4];
        #pragma unroll
        for (int mi = 0; mi < 2; ++mi) af[mi] = load_frag(arow[mi] + k0 + kb);
        #pragma unroll
        for (int ni = 0; ni < 4; ++ni) bf[ni] = load_frag(wrow[ni] + k0 + kb);
        #pragma unroll
        for (int mi = 0; mi < 2; ++mi)
            #pragma unroll
            for (int ni = 0; ni < 4; ++ni)
                acc[mi][ni] = wmma_bf16(af[mi], bf[ni], acc[mi][ni]);
    }

    // ------------------------------- epilogue -------------------------------
    #pragma unroll
    for (int ni = 0; ni < 4; ++ni) {
        const int nn = n0 + ni * 16 + ln;           // global out-feature
        if (MODE == 2) {
            float* o = (float*)dst;
            #pragma unroll
            for (int mi = 0; mi < 2; ++mi)
                #pragma unroll
                for (int v = 0; v < 8; ++v) {
                    int mr = m0 + mi * 16 + v + 8 * hi;
                    o[(size_t)mr * N + nn] = acc[mi][ni][v];
                }
        } else if (MODE == 1) {
            __bf16* o = (__bf16*)dst;
            const int h = nn >> 6, d = nn & 63;
            #pragma unroll
            for (int mi = 0; mi < 2; ++mi)
                #pragma unroll
                for (int v = 0; v < 8; ++v) {
                    int mr = m0 + mi * 16 + v + 8 * hi;
                    int b  = mr >> 11, s = mr & (SEQL - 1);
                    o[(((size_t)b * HEADS + h) * DK + d) * SEQL + s] =
                        (__bf16)acc[mi][ni][v];
                }
        } else {  // MODE 0: RoPE
            __bf16* o = (__bf16*)dst;
            const int h = nn >> 6, d = nn & 63;
            const int i2 = d & ~1;                  // 2*i of the (2i,2i+1) pair
            const float inv = __expf(-(float)i2 * (9.210340372f / 64.0f));
            const float sgn = (d & 1) ? 1.0f : -1.0f;
            #pragma unroll
            for (int mi = 0; mi < 2; ++mi)
                #pragma unroll
                for (int v = 0; v < 8; ++v) {
                    int mr = m0 + mi * 16 + v + 8 * hi;
                    int b  = mr >> 11, s = mr & (SEQL - 1);
                    float val     = acc[mi][ni][v];
                    float partner = __shfl_xor(val, 1);  // pair in adjacent lane
                    float ang = (float)pos[s] * inv;
                    float cs = cosf(ang), sn = sinf(ang);
                    // even d: x1*cos - x2*sin ; odd d: x1*sin + x2*cos
                    float out = val * cs + partner * sgn * sn;
                    o[(((size_t)b * HEADS + h) * SEQL + s) * DK + d] = (__bf16)out;
                }
        }
    }
}

// ---------------------------------------------------------------------------
// Flash-style causal attention. One wave per 16-query tile, 32-key tiles,
// online softmax. Q,K in [B,H,S,dk] bf16 (RoPE'd); V in [B,H,dk,S] bf16.
// Output Ao in [B*S, D] bf16.
__global__ __launch_bounds__(128) void attn_kernel(
    const __bf16* __restrict__ Qh, const __bf16* __restrict__ Kh,
    const __bf16* __restrict__ Vt, __bf16* __restrict__ Ao)
{
    __shared__ __bf16 lds[4][16 * 32];   // per-wave P staging (C-layout -> A-layout)

    const int lane = threadIdx.x & 31;
    const int wave = threadIdx.x >> 5;
    const int t    = blockIdx.x * 4 + wave;   // tile id over B*H*(S/16)=4096
    const int qt   = t & 127;
    const int h    = (t >> 7) & (HEADS - 1);
    const int b    = t >> 11;
    const int q0   = qt << 4;
    const int kb   = (lane & 16) ? 8 : 0;
    const int ln   = lane & 15;
    const int hi   = lane >> 4;

    const __bf16* qbase = Qh + (size_t)(b * HEADS + h) * SEQL * DK;
    const __bf16* kbase = Kh + (size_t)(b * HEADS + h) * SEQL * DK;
    const __bf16* vbase = Vt + (size_t)(b * HEADS + h) * DK * SEQL;

    // Q tile A-fragments (16x64 = two 16x32 frags), loaded once.
    const __bf16* qrow = qbase + (size_t)(q0 + ln) * DK;
    const v16bf aq0 = load_frag(qrow + kb);
    const v16bf aq1 = load_frag(qrow + 32 + kb);

    v8f acc[4] = {};                      // O tile 16x64 f32
    float mi[8], li[8];
    #pragma unroll
    for (int v = 0; v < 8; ++v) { mi[v] = -1e30f; li[v] = 0.0f; }

    __bf16* buf = lds[wave];
    const int qend = q0 + 15;

    for (int j0 = 0; j0 <= qend; j0 += 32) {
        // ---- S = Q @ K^T for keys [j0, j0+32): two 16x16 C frags ----
        v8f s0f = {}, s1f = {};
        {
            const __bf16* kr0 = kbase + (size_t)(j0 + ln) * DK;
            s0f = wmma_bf16(aq0, load_frag(kr0 + kb),      s0f);
            s0f = wmma_bf16(aq1, load_frag(kr0 + 32 + kb), s0f);
            const __bf16* kr1 = kbase + (size_t)(j0 + 16 + ln) * DK;
            s1f = wmma_bf16(aq0, load_frag(kr1 + kb),      s1f);
            s1f = wmma_bf16(aq1, load_frag(kr1 + 32 + kb), s1f);
        }

        // ---- scale, causal mask, online softmax (per C-row reductions) ----
        float alpha[8];
        #pragma unroll
        for (int v = 0; v < 8; ++v) {
            const int qr = q0 + v + 8 * hi;
            float s0 = s0f[v] * 0.125f;            // 1/sqrt(64)
            float s1 = s1f[v] * 0.125f;
            if (j0 + ln > qr)      s0 = -1e30f;
            if (j0 + 16 + ln > qr) s1 = -1e30f;
            float rmax = fmaxf(s0, s1);
            #pragma unroll
            for (int off = 1; off < 16; off <<= 1)
                rmax = fmaxf(rmax, __shfl_xor(rmax, off));   // 16-lane row group
            const float mnew = fmaxf(mi[v], rmax);
            const float a    = __expf(mi[v] - mnew);
            const float p0   = __expf(s0 - mnew);
            const float p1   = __expf(s1 - mnew);
            float rs = p0 + p1;
            #pragma unroll
            for (int off = 1; off < 16; off <<= 1)
                rs += __shfl_xor(rs, off);
            li[v]    = li[v] * a + rs;
            mi[v]    = mnew;
            alpha[v] = a;
            // stage P (C-layout) into LDS as bf16
            const int m = v + 8 * hi;
            buf[m * 32 + ln]      = (__bf16)p0;
            buf[m * 32 + 16 + ln] = (__bf16)p1;
        }
        #pragma unroll
        for (int g = 0; g < 4; ++g)
            #pragma unroll
            for (int v = 0; v < 8; ++v)
                acc[g][v] *= alpha[v];

        // order LDS stores (C-layout) before A-layout reads, same wave
        asm volatile("s_wait_dscnt 0" ::: "memory");

        // ---- O += P @ V : P as one 16x32 A-frag, V as four 32x16 B-frags ----
        const v16bf pa = load_frag(buf + ln * 32 + kb);
        #pragma unroll
        for (int g = 0; g < 4; ++g) {
            const __bf16* vr = vbase + (size_t)(g * 16 + ln) * SEQL + j0;
            acc[g] = wmma_bf16(pa, load_frag(vr + kb), acc[g]);
        }
    }

    // ---- normalize and store to Ao[B*S, D] ----
    #pragma unroll
    for (int g = 0; g < 4; ++g) {
        const int d = h * DK + g * 16 + ln;
        #pragma unroll
        for (int v = 0; v < 8; ++v) {
            const int qr = q0 + v + 8 * hi;
            Ao[((size_t)b * SEQL + qr) * DMODEL + d] = (__bf16)(acc[g][v] / li[v]);
        }
    }
}

// ---------------------------------------------------------------------------
extern "C" void kernel_launch(void* const* d_in, const int* in_sizes, int n_in,
                              void* d_out, int out_size, void* d_ws, size_t ws_size,
                              hipStream_t stream) {
    (void)in_sizes; (void)n_in; (void)out_size; (void)ws_size;

    const float* x   = (const float*)d_in[0];
    const int*   pos = (const int*)d_in[1];
    const float* Wq  = (const float*)d_in[2];
    const float* Wk  = (const float*)d_in[3];
    const float* Wv  = (const float*)d_in[4];
    const float* Wo  = (const float*)d_in[5];

    const int ROWS = BATCH * SEQL;              // 4096
    const int nX = ROWS * DMODEL;               // 4,194,304
    const int nW = DMODEL * DMODEL;             // 1,048,576

    char* ws = (char*)d_ws;
    __bf16* xb  = (__bf16*)ws; ws += (size_t)nX * 2;
    __bf16* wqb = (__bf16*)ws; ws += (size_t)nW * 2;
    __bf16* wkb = (__bf16*)ws; ws += (size_t)nW * 2;
    __bf16* wvb = (__bf16*)ws; ws += (size_t)nW * 2;
    __bf16* wob = (__bf16*)ws; ws += (size_t)nW * 2;
    __bf16* Qh  = (__bf16*)ws; ws += (size_t)nX * 2;
    __bf16* Kh  = (__bf16*)ws; ws += (size_t)nX * 2;
    __bf16* Vt  = (__bf16*)ws; ws += (size_t)nX * 2;
    __bf16* Ao  = (__bf16*)ws; ws += (size_t)nX * 2;   // total 48 MB

    f32_to_bf16_kernel<<<(nX + 255) / 256, 256, 0, stream>>>(x,  xb,  nX);
    f32_to_bf16_kernel<<<(nW + 255) / 256, 256, 0, stream>>>(Wq, wqb, nW);
    f32_to_bf16_kernel<<<(nW + 255) / 256, 256, 0, stream>>>(Wk, wkb, nW);
    f32_to_bf16_kernel<<<(nW + 255) / 256, 256, 0, stream>>>(Wv, wvb, nW);
    f32_to_bf16_kernel<<<(nW + 255) / 256, 256, 0, stream>>>(Wo, wob, nW);

    const int supertiles = (ROWS / 32) * (DMODEL / 64);   // 2048
    const dim3 gg(supertiles / 8);                        // 8 waves / block

    gemm_blocked_kernel<0><<<gg, 256, 0, stream>>>(xb, wqb, Qh, pos, ROWS, DMODEL, DMODEL);
    gemm_blocked_kernel<0><<<gg, 256, 0, stream>>>(xb, wkb, Kh, pos, ROWS, DMODEL, DMODEL);
    gemm_blocked_kernel<1><<<gg, 256, 0, stream>>>(xb, wvb, Vt, pos, ROWS, DMODEL, DMODEL);

    attn_kernel<<<BATCH * HEADS * (SEQL / 16) / 4, 128, 0, stream>>>(Qh, Kh, Vt, Ao);

    gemm_blocked_kernel<2><<<gg, 256, 0, stream>>>(Ao, wob, d_out, pos, ROWS, DMODEL, DMODEL);
}
